// Multi_Head_Self_Attention_70987219468549
// MI455X (gfx1250) — compile-verified, matrix operand order checked by reading, not run on hardware
//
#include <hip/hip_runtime.h>
#include <math.h>

typedef __attribute__((ext_vector_type(16))) _Float16 v16h;
typedef __attribute__((ext_vector_type(8)))  _Float16 v8h;
typedef __attribute__((ext_vector_type(2)))  _Float16 v2h;
typedef __attribute__((ext_vector_type(8)))  float    v8f;

#define ATT_SCALE 0.25f  // 1/sqrt(HEAD_DIM=16)

// LDS A-tile layout: t = (row*2 + hi) in [0,64), per-t stride 72 halves
// (64 data + 8 pad -> 16B-aligned, banks spread). within t: ks*16 + idx.
#define A_T_STRIDE 72
#define A_LDS_HALVES (64 * A_T_STRIDE)

// ---- monotone float <-> uint encoding for atomicMax on floats ----
__device__ inline unsigned fenc(float f) {
    unsigned u = __float_as_uint(f);
    return (u & 0x80000000u) ? ~u : (u | 0x80000000u);
}
__device__ inline float fdec(unsigned u) {
    unsigned b = (u & 0x80000000u) ? (u & 0x7FFFFFFFu) : ~u;
    return __uint_as_float(b);
}
// fenc(-inf) == 0x007FFFFF -> init value for max buffers

__device__ inline v16h ld_frag16(const _Float16* p) {   // p is 16B aligned
    v8h lo = *(const v8h*)p;
    v8h hi = *(const v8h*)(p + 8);
    return __builtin_shufflevector(lo, hi, 0,1,2,3,4,5,6,7,8,9,10,11,12,13,14,15);
}

// map global k (0..127) -> (ks, fragment idx) of the WMMA 16-bit layout
// idx = 2*j + (k&1);  j<4 covers k%32 in {0..7}, j>=4 covers {16..23}; +8 for hi half
__device__ inline void kmap(int c, int& ks, int& hi, int& idx) {
    ks = c >> 5;
    int k5 = c & 31;
    hi = (k5 >> 3) & 1;
    int lo = k5 & 7;
    int j = (k5 & 16) ? (4 + (lo >> 1)) : (lo >> 1);
    idx = (j << 1) | (c & 1);
}

// ---- one-time weight pack: P[((col*2+hi)*4+ks)*16+idx] = (f16) W[col*128+k] ----
__global__ void pack_w_kernel(const float* __restrict__ W, _Float16* __restrict__ P) {
    int i = blockIdx.x * blockDim.x + threadIdx.x;   // 0..16383
    int idx = i & 15, ks = (i >> 4) & 3, hi = (i >> 6) & 1, col = i >> 7;
    int j = idx >> 1;
    int kj = ((j < 4) ? (j << 1) : (16 + ((j - 4) << 1)))
             + (hi << 3) + (ks << 5) + (idx & 1);
    P[i] = (_Float16)W[col * 128 + kj];
}

// ---- stage 32 rows of A (f32 global) into swizzled f16 LDS ----
__device__ inline void stage_A(const float* __restrict__ A, _Float16* sAh,
                               int row0, int M, int tid) {
#pragma unroll
    for (int i = 0; i < 16; ++i) {                   // 4096 f32-pairs / 256 thr
        int p = tid + (i << 8);
        int m = p >> 6, c = (p & 63) << 1;
        int gr = row0 + m;
        float2 val = make_float2(0.f, 0.f);
        if (gr < M) val = *(const float2*)(A + (size_t)gr * 128 + c);
        int ks, hi, idx;
        kmap(c, ks, hi, idx);                        // idx is even here
        _Float16* dp = sAh + ((m << 1) | hi) * A_T_STRIDE + (ks << 4) + idx;
        v2h h; h[0] = (_Float16)val.x; h[1] = (_Float16)val.y;
        *(v2h*)dp = h;                               // 4B-aligned packed store
    }
}

// ---- per-wave: two 16x16 row-tiles x 128 cols, K=128 ----
__device__ inline void wmma_2tiles(const _Float16* sAh, const _Float16* __restrict__ Wp,
                                   int ncol, int lane, v8f& acc0, v8f& acc1) {
    const int mrow = lane & 15, hi = lane >> 4;
    const _Float16* a0p = sAh + (((mrow) << 1) | hi) * A_T_STRIDE;
    const _Float16* a1p = sAh + (((mrow + 16) << 1) | hi) * A_T_STRIDE;
    const _Float16* bp  = Wp + (((size_t)ncol << 1) | hi) * 64;
#pragma unroll
    for (int ks = 0; ks < 4; ++ks) {
        v16h b  = ld_frag16(bp + (ks << 4));         // global, pre-packed f16
        v16h a0 = ld_frag16(a0p + (ks << 4));        // LDS, pre-swizzled f16
        v16h a1 = ld_frag16(a1p + (ks << 4));
        acc0 = __builtin_amdgcn_wmma_f32_16x16x32_f16(false, a0, false, b,
                                                      (short)0, acc0, false, false);
        acc1 = __builtin_amdgcn_wmma_f32_16x16x32_f16(false, a1, false, b,
                                                      (short)0, acc1, false, false);
    }
}

// ---- generic 128-wide projection: out = act(A @ W.T + bias), 32 rows/block ----
template <int ACT>
__global__ void __launch_bounds__(256)
gemm128_kernel(const float* __restrict__ A, const _Float16* __restrict__ Wp,
               const float* __restrict__ bias, float* __restrict__ out, int M) {
    __shared__ _Float16 sAh[A_LDS_HALVES];
    const int tid = threadIdx.x;
    const int row0 = blockIdx.x << 5;
    stage_A(A, sAh, row0, M, tid);
    __syncthreads();
    const int wave = tid >> 5, lane = tid & 31, hi = lane >> 4;
    const int ncol = (wave << 4) + (lane & 15);
    v8f acc0 = {}, acc1 = {};
    wmma_2tiles(sAh, Wp, ncol, lane, acc0, acc1);
    const float bv = bias[ncol];
#pragma unroll
    for (int r = 0; r < 8; ++r) {
        int m0 = r + (hi << 3);
        int gr0 = row0 + m0, gr1 = gr0 + 16;
        float v0 = acc0[r] + bv, v1 = acc1[r] + bv;
        if (ACT) {
            v0 = (v0 > 0.f) ? v0 : 0.1f * v0;        // leaky_relu(0.1)
            v1 = (v1 > 0.f) ? v1 : 0.1f * v1;
        }
        if (gr0 < M) out[(size_t)gr0 * 128 + ncol] = v0;
        if (gr1 < M) out[(size_t)gr1 * 128 + ncol] = v1;
    }
}

// ---- fused: k = edge_h @ Wk.T + bk (kept in LDS), then 3 per-head logits
//      + atomic segment-max per (dst, head, type).  32 edges/block. ----
__global__ void __launch_bounds__(256)
k_logits_kernel(const float* __restrict__ edge_h, const _Float16* __restrict__ Wkp,
                const float* __restrict__ bk, const float* __restrict__ q,
                const int* __restrict__ src, const int* __restrict__ dst,
                float* __restrict__ logits, unsigned* __restrict__ maxenc, int E) {
    __shared__ _Float16 sAh[A_LDS_HALVES];
    __shared__ float sK[32][132];
    const int tid = threadIdx.x;
    const int row0 = blockIdx.x << 5;
    stage_A(edge_h, sAh, row0, E, tid);
    __syncthreads();
    const int wave = tid >> 5, lane = tid & 31, hi = lane >> 4;
    const int ncol = (wave << 4) + (lane & 15);
    v8f acc0 = {}, acc1 = {};
    wmma_2tiles(sAh, Wkp, ncol, lane, acc0, acc1);
    const float bv = bk[ncol];
#pragma unroll
    for (int r = 0; r < 8; ++r) {
        int m0 = r + (hi << 3);
        sK[m0][ncol]      = acc0[r] + bv;            // k rows never touch HBM
        sK[m0 + 16][ncol] = acc1[r] + bv;
    }
    __syncthreads();
    {                                                // (32 edges) x (8 heads)
        const int el = tid >> 3, head = tid & 7;
        const int e = row0 + el;
        if (e < E) {
            const int s = src[e], d = dst[e];
            const float* qs = q + (size_t)s * 128 + head * 16;
            const float* qd = q + (size_t)d * 128 + head * 16;
            const float* kk = &sK[el][head * 16];
            float m_in = 0.f, m_out = 0.f, m_dg = 0.f;
#pragma unroll
            for (int i = 0; i < 16; ++i) {
                float a = qs[i], b = qd[i], c = kk[i];
                m_in += a * c; m_out += b * c; m_dg += a * b;
            }
            const float l0 = m_in * ATT_SCALE, l1 = m_out * ATT_SCALE,
                        l2 = m_dg * ATT_SCALE;
            const size_t lb = (size_t)e * 24 + head * 3;
            logits[lb] = l0; logits[lb + 1] = l1; logits[lb + 2] = l2;
            unsigned* mp = maxenc + (size_t)d * 24 + head * 3;
            atomicMax(&mp[0], fenc(l0));
            atomicMax(&mp[1], fenc(l1));
            atomicMax(&mp[2], fenc(l2));
        }
    }
}

__global__ void init_kernel(unsigned* __restrict__ maxenc, float* __restrict__ num,
                            float* __restrict__ den, int n) {
    int i = blockIdx.x * blockDim.x + threadIdx.x;
    if (i < n) { maxenc[i] = 0x007FFFFFu; num[i] = 0.f; den[i] = 0.f; }
}

// ---- per (edge,head): e_t = exp(logit - max); den += e_t; num += e_t*att ----
__global__ void exp_kernel(const float* __restrict__ logits,
                           const unsigned* __restrict__ maxenc,
                           const float* __restrict__ distance,
                           const float* __restrict__ lamp,
                           const int* __restrict__ dst,
                           float* __restrict__ num, float* __restrict__ den, int E) {
    int idx = blockIdx.x * blockDim.x + threadIdx.x;
    if (idx >= E * 8) return;
    const int e = idx >> 3, head = idx & 7;
    const int d = dst[e];
    const float att = powf(distance[e], *lamp);
    const size_t lb = (size_t)e * 24 + head * 3;
    const size_t nb = (size_t)d * 24 + head * 3;
#pragma unroll
    for (int t = 0; t < 3; ++t) {
        float m = fdec(maxenc[nb + t]);
        if (!(m > -3.0e38f)) m = 0.f;                // where(isfinite(m), m, 0)
        float ex = expf(logits[lb + t] - m);
        atomicAdd(&den[nb + t], ex);
        atomicAdd(&num[nb + t], ex * att);
    }
}

// ---- h[d] = v[d] * S[d,head],  S = sum_t num/den (den==0 -> 1) ----
__global__ void combine_kernel(const float* __restrict__ v,
                               const float* __restrict__ num,
                               const float* __restrict__ den,
                               float* __restrict__ h, int N) {
    int idx = blockIdx.x * blockDim.x + threadIdx.x;
    if (idx >= N * 128) return;
    const int d = idx >> 7, c = idx & 127, head = c >> 4;
    const size_t nb = (size_t)d * 24 + head * 3;
    float S = 0.f;
#pragma unroll
    for (int t = 0; t < 3; ++t) {
        float dn = den[nb + t];
        dn = (dn == 0.f) ? 1.f : dn;
        S += num[nb + t] / dn;
    }
    h[idx] = v[idx] * S;
}

extern "C" void kernel_launch(void* const* d_in, const int* in_sizes, int n_in,
                              void* d_out, int out_size, void* d_ws, size_t ws_size,
                              hipStream_t stream) {
    const float* node_h   = (const float*)d_in[0];
    const float* edge_h   = (const float*)d_in[1];
    const float* distance = (const float*)d_in[2];
    const float* Wq = (const float*)d_in[3];
    const float* bq = (const float*)d_in[4];
    const float* Wk = (const float*)d_in[5];
    const float* bk = (const float*)d_in[6];
    const float* Wv = (const float*)d_in[7];
    const float* bv = (const float*)d_in[8];
    const float* Wo = (const float*)d_in[9];
    const float* bo = (const float*)d_in[10];
    const float* lam = (const float*)d_in[11];
    const int* src = (const int*)d_in[12];
    const int* dst = (const int*)d_in[13];
    const int N = in_sizes[0] / 128;
    const int E = in_sizes[12];
    (void)n_in; (void)out_size; (void)ws_size;

    char* ws = (char*)d_ws;
    size_t off = 0;
    auto carve = [&](size_t bytes) -> void* {
        void* p = ws + off;
        off += (bytes + 255) & ~(size_t)255;
        return p;
    };
    float*    qbuf   = (float*)carve((size_t)N * 128 * 4);
    float*    vbuf   = (float*)carve((size_t)N * 128 * 4);
    float*    hbuf   = (float*)carve((size_t)N * 128 * 4);
    float*    logits = (float*)carve((size_t)E * 24 * 4);
    unsigned* maxenc = (unsigned*)carve((size_t)N * 24 * 4);
    float*    num    = (float*)carve((size_t)N * 24 * 4);
    float*    den    = (float*)carve((size_t)N * 24 * 4);
    _Float16* Wqp    = (_Float16*)carve((size_t)16384 * 2);
    _Float16* Wkp    = (_Float16*)carve((size_t)16384 * 2);
    _Float16* Wvp    = (_Float16*)carve((size_t)16384 * 2);
    _Float16* Wop    = (_Float16*)carve((size_t)16384 * 2);

    // one-time weight packs (f32 row-major -> f16 WMMA B-fragment layout)
    pack_w_kernel<<<64, 256, 0, stream>>>(Wq, Wqp);
    pack_w_kernel<<<64, 256, 0, stream>>>(Wk, Wkp);
    pack_w_kernel<<<64, 256, 0, stream>>>(Wv, Wvp);
    pack_w_kernel<<<64, 256, 0, stream>>>(Wo, Wop);

    const int n24 = N * 24;
    init_kernel<<<(n24 + 255) / 256, 256, 0, stream>>>(maxenc, num, den, n24);

    gemm128_kernel<0><<<(N + 31) / 32, 256, 0, stream>>>(node_h, Wqp, bq, qbuf, N);
    gemm128_kernel<0><<<(N + 31) / 32, 256, 0, stream>>>(node_h, Wvp, bv, vbuf, N);

    k_logits_kernel<<<(E + 31) / 32, 256, 0, stream>>>(edge_h, Wkp, bk, qbuf,
                                                       src, dst, logits, maxenc, E);

    exp_kernel<<<(E * 8 + 255) / 256, 256, 0, stream>>>(logits, maxenc, distance,
                                                        lam, dst, num, den, E);

    combine_kernel<<<(N * 128 + 255) / 256, 256, 0, stream>>>(vbuf, num, den, hbuf, N);

    gemm128_kernel<1><<<(N + 31) / 32, 256, 0, stream>>>(hbuf, Wop, bo,
                                                         (float*)d_out, N);
}
